// QuantumInspiredAttention_68470368633636
// MI455X (gfx1250) — compile-verified, matrix operand order checked by reading, not run on hardware
//
#include <hip/hip_runtime.h>
#include <hip/hip_bf16.h>
#include <math.h>

// ---------------------------------------------------------------------------
// QuantumInspiredAttention for MI455X (gfx1250), wave32 + WMMA bf16.
//
// Algebraic collapse: ent->qv->m1 is affine in x, so
//   pre[b,t,f] = sum_d x[b,t,d] * M1[f,d] + c[b,f]
// M1 is folded once (WMMA) and stored in WMMA B-fragment-ready layout (M1F)
// so the main GEMM loads B operands with coalesced b128 loads from L2.
// Then h = tanh(LN(pre)), logit = h . m2_w, softmax over T, context = w . x.
// ---------------------------------------------------------------------------

#define Dq 512
#define Sq 8
#define Bq 16
#define Tq 1024
#define Gq 4096   // D*S
#define LN_EPS 1e-5f

typedef __attribute__((ext_vector_type(16))) __bf16 v16bf;
typedef __attribute__((ext_vector_type(8)))  float  v8f;

struct BF16x16 { unsigned short u[16]; };
struct Frag32  { uint4 lo, hi; };                  // 32 bytes = one v16bf fragment

__device__ __forceinline__ unsigned short f2bf(float f) {
    unsigned int u = __float_as_uint(f);
    unsigned int r = u + 0x7fffu + ((u >> 16) & 1u);   // round-to-nearest-even
    return (unsigned short)(r >> 16);
}
__device__ __forceinline__ unsigned int pack_bf2(float a, float b) {
    return (unsigned int)f2bf(a) | ((unsigned int)f2bf(b) << 16);
}

// M1F fragment indexing: frag (ft, kb, lane) holds the 16 bf16 B-elements for
// f-tile ft (16 n-cols), K-step kb (32 k), in ISA 16-bit operand layout:
//   element e of lane L: k = (e/8)*16 + (L/16)*8 + (e%8),  n = L%16
__device__ __forceinline__ size_t m1f_idx(int ft, int kb, int lane2, int e) {
    return ((size_t)((ft * 16 + kb) * 32 + lane2) << 4) + e;
}

// ---------------------------------------------------------------------------
// Kernel 1: xbar[b,d] = mean_t x[b,t,d]          (B*D threads)
// ---------------------------------------------------------------------------
__global__ __launch_bounds__(256) void xbar_kernel(const float* __restrict__ x,
                                                   float* __restrict__ xbar) {
    int idx = blockIdx.x * 256 + threadIdx.x;      // 8192
    int b = idx >> 9, d = idx & 511;
    const float* p = x + (size_t)b * Tq * Dq + d;
    float s = 0.f;
    for (int t = 0; t < Tq; ++t) s += p[(size_t)t * Dq];
    xbar[idx] = s * (1.0f / Tq);
}

// ---------------------------------------------------------------------------
// Kernel 2: y[b,g] = xbar[b]·W2[g,:] + b_ent[g]   (g = e*S+s ; B*G threads)
// ---------------------------------------------------------------------------
__global__ __launch_bounds__(256) void y_kernel(const float* __restrict__ xbar,
                                                const float* __restrict__ W_ent,
                                                const float* __restrict__ b_ent,
                                                float* __restrict__ y) {
    int idx = blockIdx.x * 256 + threadIdx.x;      // 65536
    int b = idx >> 12, g = idx & 4095;
    int s = g & 7, e = g >> 3;
    const float* w  = W_ent + ((size_t)(s * Dq + e)) * (2 * Dq) + Dq;
    const float* xb = xbar + b * Dq;
    float acc = b_ent[s * Dq + e];
    for (int d = 0; d < Dq; ++d) acc += xb[d] * w[d];
    y[idx] = acc;
}

// ---------------------------------------------------------------------------
// Kernel 3: c[b,f] = m1_w[f,:]·y[b,:] + m1_b[f]   (B*D threads)
// ---------------------------------------------------------------------------
__global__ __launch_bounds__(256) void c_kernel(const float* __restrict__ y,
                                                const float* __restrict__ m1_w,
                                                const float* __restrict__ m1_b,
                                                float* __restrict__ c) {
    int idx = blockIdx.x * 256 + threadIdx.x;      // 8192
    int b = idx >> 9, f = idx & 511;
    const float* yr = y + b * Gq;
    const float* wr = m1_w + (size_t)f * Gq;
    float acc = m1_b[f];
    for (int g = 0; g < Gq; ++g) acc += yr[g] * wr[g];
    c[idx] = acc;
}

// ---------------------------------------------------------------------------
// Kernel 4: fold  M1[f,d] = sum_g W1[g,d] * m1_w[f,g]  -> M1F (frag layout)
// WMMA: per wave one 16(d) x 16(f) tile, K = 4096 in steps of 32.
// ---------------------------------------------------------------------------
__global__ __launch_bounds__(256) void fold_kernel(const float* __restrict__ W_ent,
                                                   const float* __restrict__ m1_w,
                                                   unsigned short* __restrict__ M1F) {
    const int tid = threadIdx.x, lane = tid & 31, wave = tid >> 5;
    const int wg = blockIdx.x * 8 + wave;          // 0..1023 tiles
    const int d0 = (wg >> 5) << 4;
    const int f0 = (wg & 31) << 4;
    const int mn = lane & 15, hi = lane >> 4;

    v8f acc = {};
    for (int kb = 0; kb < Gq / 32; ++kb) {
        BF16x16 aS, bS;
#pragma unroll
        for (int j = 0; j < 8; ++j) {
            // CDNA5 16-bit A-matrix VGPR k-mapping (ISA 7.12.2)
            int k0 = (j >> 2) * 16 + hi * 8 + ((j & 3) << 1);
            int g0 = kb * 32 + k0, g1 = g0 + 1;
            // A: W1[g, d0+m] = W_ent[(g%S)*D + g/S][d]
            aS.u[2 * j]     = f2bf(W_ent[(size_t)((g0 & 7) * Dq + (g0 >> 3)) * (2 * Dq) + d0 + mn]);
            aS.u[2 * j + 1] = f2bf(W_ent[(size_t)((g1 & 7) * Dq + (g1 >> 3)) * (2 * Dq) + d0 + mn]);
            // B: m1_w[f0+n, g]
            bS.u[2 * j]     = f2bf(m1_w[(size_t)(f0 + mn) * Gq + g0]);
            bS.u[2 * j + 1] = f2bf(m1_w[(size_t)(f0 + mn) * Gq + g1]);
        }
        v16bf a  = __builtin_bit_cast(v16bf, aS);
        v16bf bb = __builtin_bit_cast(v16bf, bS);
        acc = __builtin_amdgcn_wmma_f32_16x16x32_bf16(false, a, false, bb,
                                                      (short)0, acc, false, false);
    }
    // Scatter accumulator directly into B-fragment-ready layout.
    // This lane holds M1[d0 + r + 8*hi (d), f0 + mn (f)] in acc[r].
#pragma unroll
    for (int r = 0; r < 8; ++r) {
        int d   = d0 + r + 8 * hi;
        int f   = f0 + mn;
        int kb2 = d >> 5;
        int kin = d & 31;
        int hi2 = (kin >> 3) & 1;                  // which half-wave of lanes
        int e   = ((kin >> 4) << 3) + (kin & 7);   // fragment element index
        int ln2 = hi2 * 16 + (f & 15);
        M1F[m1f_idx(f >> 4, kb2, ln2, e)] = f2bf(acc[r]);
    }
}

// ---------------------------------------------------------------------------
// Kernel 5 (main): per block = 16 t-rows of one b, all 512 f.
//   512 threads = 16 waves; wave w owns f-tiles 2w, 2w+1.
//   A frags: 2x ds_load_b128 from bf16 x block in LDS (ISA A-layout runs).
//   B frags: 2x global_load_b128 per tile from fragment-ready M1F (L2-hot).
//   After GEMM: LN + tanh + dot(m2_w) per row (one wave per row, shfl reduce).
// ---------------------------------------------------------------------------
__global__ __launch_bounds__(512) void main_gemm_kernel(const float* __restrict__ x,
                                                        const unsigned short* __restrict__ M1F,
                                                        const float* __restrict__ c,
                                                        const float* __restrict__ ln_g,
                                                        const float* __restrict__ ln_b,
                                                        const float* __restrict__ m2_w,
                                                        const float* __restrict__ m2_b,
                                                        float* __restrict__ logits) {
    __shared__ unsigned short xs[16 * Dq];         // bf16 x block [t][d]   (16KB)
    __shared__ float pre[16 * Dq];                 // pre-LN activations    (32KB)

    const int tid = threadIdx.x, lane = tid & 31, wave = tid >> 5;
    const int blk = blockIdx.x;                    // 1024 blocks
    const int b = blk >> 6;                        // 64 t-blocks per batch
    const int t0 = (blk & 63) << 4;
    const int mn = lane & 15, hi = lane >> 4;
    const int f0 = wave << 5;                      // two tiles: f0, f0+16

    // Stage x block (contiguous 16x512 f32) -> bf16 LDS
    {
        const float4* src = reinterpret_cast<const float4*>(x + ((size_t)b * Tq + t0) * Dq);
        unsigned int* xd = reinterpret_cast<unsigned int*>(xs);
#pragma unroll
        for (int i = 0; i < 4; ++i) {
            float4 v = src[tid * 4 + i];
            xd[tid * 8 + 2 * i]     = pack_bf2(v.x, v.y);
            xd[tid * 8 + 2 * i + 1] = pack_bf2(v.z, v.w);
        }
    }
    __syncthreads();

    // Per-lane fragment base pointers into M1F (uint4 units; 2 uint4 per frag).
    // frag(ft, kb, lane) at uint4 offset ((ft*16 + kb)*32 + lane)*2
    const uint4* m1f4 = reinterpret_cast<const uint4*>(M1F);
    const uint4* b0p = m1f4 + ((size_t)((2 * wave) * 16) * 32 + lane) * 2;
    const uint4* b1p = m1f4 + ((size_t)((2 * wave + 1) * 16) * 32 + lane) * 2;

    v8f acc0 = {}, acc1 = {};
    for (int kb = 0; kb < Dq / 32; ++kb) {
        // A fragment: two contiguous 16B runs per lane (ISA 16-bit A layout)
        const uint4* ar = reinterpret_cast<const uint4*>(xs + mn * Dq + kb * 32 + hi * 8);
        Frag32 aF { ar[0], ar[2] };                // k = +0..7 and +16..23 (per hi)
        // B fragments: contiguous 32B per lane, coalesced 1KB per wave
        Frag32 b0F { b0p[kb * 64], b0p[kb * 64 + 1] };
        Frag32 b1F { b1p[kb * 64], b1p[kb * 64 + 1] };
        if (kb + 1 < Dq / 32)                      // -> global_prefetch_b8
            __builtin_prefetch((const void*)(b0p + (kb + 1) * 64), 0, 0);

        v16bf a   = __builtin_bit_cast(v16bf, aF);
        v16bf bb0 = __builtin_bit_cast(v16bf, b0F);
        v16bf bb1 = __builtin_bit_cast(v16bf, b1F);
        acc0 = __builtin_amdgcn_wmma_f32_16x16x32_bf16(false, a, false, bb0,
                                                       (short)0, acc0, false, false);
        acc1 = __builtin_amdgcn_wmma_f32_16x16x32_bf16(false, a, false, bb1,
                                                       (short)0, acc1, false, false);
    }

    // Write pre-LN tile (+ folded bias c[b,f])
    {
        const float* crow = c + b * Dq;
        float c0 = crow[f0 + mn], c1 = crow[f0 + 16 + mn];
#pragma unroll
        for (int r = 0; r < 8; ++r) {
            int mm = r + 8 * hi;                   // C/D layout: row = r + 8*(lane/16)
            pre[mm * Dq + f0 + mn]      = acc0[r] + c0;
            pre[mm * Dq + f0 + 16 + mn] = acc1[r] + c1;
        }
    }
    __syncthreads();

    // LayerNorm + tanh + dot(m2_w): wave w handles row t = t0 + w
    const float* row = pre + wave * Dq;
    float sum = 0.f, sq = 0.f;
    for (int f = lane; f < Dq; f += 32) { float v = row[f]; sum += v; sq += v * v; }
#pragma unroll
    for (int off = 16; off; off >>= 1) {
        sum += __shfl_xor(sum, off, 32);
        sq  += __shfl_xor(sq,  off, 32);
    }
    float mu  = sum * (1.0f / Dq);
    float var = sq * (1.0f / Dq) - mu * mu;
    float inv = rsqrtf(var + LN_EPS);
    float dot = 0.f;
    for (int f = lane; f < Dq; f += 32) {
        float h = tanhf((row[f] - mu) * inv * ln_g[f] + ln_b[f]);
        dot += h * m2_w[f];
    }
#pragma unroll
    for (int off = 16; off; off >>= 1) dot += __shfl_xor(dot, off, 32);
    if (lane == 0) logits[b * Tq + t0 + wave] = dot + m2_b[0];
}

// ---------------------------------------------------------------------------
// Kernel 6: softmax over T per batch -> attn weights (written to d_out tail)
// ---------------------------------------------------------------------------
__global__ __launch_bounds__(256) void softmax_kernel(const float* __restrict__ logits,
                                                      float* __restrict__ attn) {
    __shared__ float red[256];
    int b = blockIdx.x, tid = threadIdx.x;
    const float* l = logits + b * Tq;
    float mx = -1e30f;
    for (int t = tid; t < Tq; t += 256) mx = fmaxf(mx, l[t]);
    red[tid] = mx; __syncthreads();
    for (int s = 128; s; s >>= 1) { if (tid < s) red[tid] = fmaxf(red[tid], red[tid + s]); __syncthreads(); }
    mx = red[0]; __syncthreads();
    float sum = 0.f;
    for (int t = tid; t < Tq; t += 256) sum += __expf(l[t] - mx);
    red[tid] = sum; __syncthreads();
    for (int s = 128; s; s >>= 1) { if (tid < s) red[tid] += red[tid + s]; __syncthreads(); }
    float invs = 1.0f / red[0];
    for (int t = tid; t < Tq; t += 256) attn[b * Tq + t] = __expf(l[t] - mx) * invs;
}

// ---------------------------------------------------------------------------
// Kernel 7: context[b,d] = sum_t attn[b,t] * x[b,t,d]
// ---------------------------------------------------------------------------
__global__ __launch_bounds__(256) void context_kernel(const float* __restrict__ x,
                                                      const float* __restrict__ attn,
                                                      float* __restrict__ ctx) {
    int idx = blockIdx.x * 256 + threadIdx.x;      // 8192
    int b = idx >> 9, d = idx & 511;
    const float* w = attn + b * Tq;
    const float* p = x + (size_t)b * Tq * Dq + d;
    float s = 0.f;
    for (int t = 0; t < Tq; ++t) s += w[t] * p[(size_t)t * Dq];
    ctx[idx] = s;
}

// ---------------------------------------------------------------------------
extern "C" void kernel_launch(void* const* d_in, const int* in_sizes, int n_in,
                              void* d_out, int out_size, void* d_ws, size_t ws_size,
                              hipStream_t stream) {
    const float* x     = (const float*)d_in[0];
    const float* W_ent = (const float*)d_in[1];
    const float* b_ent = (const float*)d_in[2];
    const float* m1_w  = (const float*)d_in[3];
    const float* m1_b  = (const float*)d_in[4];
    const float* ln_g  = (const float*)d_in[5];
    const float* ln_b  = (const float*)d_in[6];
    const float* m2_w  = (const float*)d_in[7];
    const float* m2_b  = (const float*)d_in[8];

    char* ws = (char*)d_ws;
    float*          xbar   = (float*)(ws + 0);               //  32 KB
    float*          y      = (float*)(ws + 32768);           // 256 KB
    float*          c      = (float*)(ws + 294912);          //  32 KB
    unsigned short* M1F    = (unsigned short*)(ws + 327680); // 512 KB (bf16 frags)
    float*          logits = (float*)(ws + 851968);          //  64 KB

    float* ctx  = (float*)d_out;                 // context: B*D = 8192 floats
    float* attn = (float*)d_out + Bq * Dq;       // attn_weights: B*T = 16384 floats

    xbar_kernel     <<<  32, 256, 0, stream>>>(x, xbar);
    fold_kernel     <<< 128, 256, 0, stream>>>(W_ent, m1_w, M1F);
    y_kernel        <<< 256, 256, 0, stream>>>(xbar, W_ent, b_ent, y);
    c_kernel        <<<  32, 256, 0, stream>>>(y, m1_w, m1_b, c);
    main_gemm_kernel<<<1024, 512, 0, stream>>>(x, M1F, c, ln_g, ln_b, m2_w, m2_b, logits);
    softmax_kernel  <<<  Bq, 256, 0, stream>>>(logits, attn);
    context_kernel  <<<  32, 256, 0, stream>>>(x, attn, ctx);
}
